// SimpleAttention_31576599560710
// MI455X (gfx1250) — matrix operand order repaired and checked by reading, not compiled, not run on hardware
//
#include <hip/hip_runtime.h>

// ---------------------------------------------------------------------------
// CDNA5 (gfx1250) wave32 WMMA attention block.
// All matrix math on v_wmma_f32_16x16x32_f16 (f16 operands, f32 accum).
// V tiles in the flash-attention kernel are staged by the Tensor Data Mover
// (tensor_load_to_lds + s_wait_tensorcnt) when the builtin is available.
// ---------------------------------------------------------------------------

typedef __attribute__((ext_vector_type(16))) _Float16 v16h;
typedef __attribute__((ext_vector_type(8)))  float    v8f;
typedef unsigned int u32x4 __attribute__((ext_vector_type(4)));
typedef int          i32x4 __attribute__((ext_vector_type(4)));
typedef int          i32x8 __attribute__((ext_vector_type(8)));

#define WMMA_F16(a, b, c) \
  __builtin_amdgcn_wmma_f32_16x16x32_f16(false, (a), false, (b), (short)0, (c), false, false)

#if __has_builtin(__builtin_amdgcn_tensor_load_to_lds) && \
    __has_builtin(__builtin_amdgcn_s_wait_tensorcnt)
#define USE_TDM 1
#else
#define USE_TDM 0
#endif

// A-matrix (16x32, f16) element->K mapping for lane `l`, vector element `e`:
//   lanes 0-15 hold M=0..15; K = e + (lane<16 ? 0 : 16)
__device__ __forceinline__ int a_k(int e, int lane) {
  return e + (lane & 16);
}
// B-matrix (32x16, f16): lane holds column n = lane&15; K = e + (lane<16 ? 0 : 16)
__device__ __forceinline__ int b_k(int e, int lane) {
  return e + (lane & 16);
}

// ---------------------------------------------------------------------------
// f32 -> f16 convert
// ---------------------------------------------------------------------------
__global__ void cvt_f16(const float* __restrict__ in, _Float16* __restrict__ out, int n) {
  int i = blockIdx.x * blockDim.x + threadIdx.x;
  if (i < n) out[i] = (_Float16)in[i];
}

// ---------------------------------------------------------------------------
// RoPE cos/sin tables: [2048][64] cos followed by [2048][64] sin (f32).
// Keeps the heavy transcendental expansion out of the GEMM epilogue.
// ---------------------------------------------------------------------------
__global__ void rope_tables(float* __restrict__ cs) {
  int i = blockIdx.x * blockDim.x + threadIdx.x;   // 2048*64 entries
  if (i >= 2048 * 64) return;
  const int n = i >> 6, hd = i & 63;
  const float inv_freq = powf(10000.0f, -(float)(hd & ~1) / 64.0f);
  const float th = (float)n * inv_freq;
  cs[i]             = cosf(th);
  cs[i + 2048 * 64] = sinf(th);
}

// ---------------------------------------------------------------------------
// QKV GEMM: [4096,1024] x [3072,1024]^T + bias, RoPE on Q/K (table lookup),
// scale folded into Q, scatter to Q/K/V buffers in [B,H,N,64] f16 layout.
// Block tile 128x64, K-chunk 32, 8 waves (each: 1 A-frag, 4 B-frags, 4 wmma).
// ---------------------------------------------------------------------------
__global__ __launch_bounds__(256) void qkv_gemm_rope(
    const _Float16* __restrict__ A,   // x  [4096,1024]
    const _Float16* __restrict__ W,   // qkv_w [3072,1024]
    const float*    __restrict__ bias,
    const float*    __restrict__ ropeCS,   // cos table; sin at +131072
    _Float16* __restrict__ Qg, _Float16* __restrict__ Kg, _Float16* __restrict__ Vg) {
  constexpr int BM = 128, BN = 64, BK = 32, LP = 40;
  __shared__ __align__(16) _Float16 As[BM][LP];
  __shared__ __align__(16) _Float16 Bs[BN][LP];
  const int tid = threadIdx.x;
  const int lane = tid & 31, wave = tid >> 5;
  const int tm = blockIdx.x, tn = blockIdx.y;
  const int K = 1024;

  v8f acc[4] = {};

  for (int k0 = 0; k0 < K; k0 += BK) {
    __syncthreads();
    {   // A tile: 128 rows x 32, 2 threads/row x 16 halves
      const int row = tid >> 1, ks = (tid & 1) * 16;
      const _Float16* src = A + (size_t)(tm * BM + row) * K + k0 + ks;
      #pragma unroll
      for (int i = 0; i < 16; ++i) As[row][ks + i] = src[i];
      if (k0 + BK < K) __builtin_prefetch(src + BK, 0, 1);
    }
    {   // B tile: 64 rows x 32, 4 threads/row x 8 halves
      const int row = tid >> 2, ks = (tid & 3) * 8;
      const _Float16* src = W + (size_t)(tn * BN + row) * K + k0 + ks;
      #pragma unroll
      for (int i = 0; i < 8; ++i) Bs[row][ks + i] = src[i];
    }
    __syncthreads();

    v16h a;
    #pragma unroll
    for (int e = 0; e < 16; ++e) a[e] = As[wave * 16 + (lane & 15)][a_k(e, lane)];
    #pragma unroll
    for (int nt = 0; nt < 4; ++nt) {
      v16h b;
      #pragma unroll
      for (int e = 0; e < 16; ++e) b[e] = Bs[nt * 16 + (lane & 15)][b_k(e, lane)];
      acc[nt] = WMMA_F16(a, b, acc[nt]);
    }
  }

  // Epilogue: bias, RoPE (interleaved pairs via lane^1 shuffle), scale, scatter.
  #pragma unroll
  for (int nt = 0; nt < 4; ++nt) {
    const int j = tn * 64 + nt * 16 + (lane & 15);   // output column in [0,3072)
    const float bj = bias[j];
    const int sel = j >> 10;                          // 0=Q 1=K 2=V (uniform per subtile)
    const int hc = j & 1023, head = hc >> 6, hd = hc & 63;
    #pragma unroll
    for (int r = 0; r < 8; ++r) {
      const int mg = tm * 128 + wave * 16 + ((lane & 16) >> 1) + r;  // global row
      const int n = mg & 2047, bb = mg >> 11;
      float v = acc[nt][r] + bj;
      const float partner = __shfl_xor(v, 1, 32);     // value at hd^1
      if (sel < 2) {
        const int ti = n * 64 + hd;
        const float ct = ropeCS[ti];
        const float st = ropeCS[131072 + ti];
        const float rot = (hd & 1) ? partner : -partner;
        v = v * ct + rot * st;
        if (sel == 0) v *= 0.125f;                    // fold 1/sqrt(64) into Q
      }
      const size_t idx = (((size_t)bb * 16 + head) * 2048 + n) * 64 + hd;
      const _Float16 hv = (_Float16)v;
      if (sel == 0)      Qg[idx] = hv;
      else if (sel == 1) Kg[idx] = hv;
      else               Vg[idx] = hv;
    }
  }
}

// ---------------------------------------------------------------------------
// Flash attention: one wave per (b,h,16-row tile). Keys in chunks of 32.
// S = Q K^T via 2 chained wmma per 16x16 tile; online softmax; P routed
// through LDS to A-layout; O += P V via 4 wmma per chunk.
// V tile (32 x 128B) staged by TDM into bank-skewed LDS (pitch 144B) when
// tensor_load_to_lds is available; overlapped with score computation.
// ---------------------------------------------------------------------------
__global__ __launch_bounds__(32) void attn_fwd(
    const _Float16* __restrict__ Qg, const _Float16* __restrict__ Kg,
    const _Float16* __restrict__ Vg, _Float16* __restrict__ attnb) {
  __shared__ __align__(16) _Float16 P[16][40];    // 16 rows x 32 keys
  __shared__ __align__(16) _Float16 Vs[32][72];   // 32 keys x 64 dims, pitch 144B
  const int lane = threadIdx.x;
  const int it = blockIdx.x;            // row tile 0..127
  const int bh = blockIdx.y;            // 0..31
  const int b = bh >> 4, h = bh & 15;
  const int i0 = it * 16;
  const size_t bhoff = (size_t)bh * 2048 * 64;

  // Q tile -> two A fragments (d 0..31, 32..63)
  v16h qa0, qa1;
  {
    const _Float16* qrow = Qg + bhoff + (size_t)(i0 + (lane & 15)) * 64;
    #pragma unroll
    for (int e = 0; e < 16; ++e) {
      qa0[e] = qrow[a_k(e, lane)];
      qa1[e] = qrow[32 + a_k(e, lane)];
    }
  }

#if USE_TDM
  const unsigned lds_vs = (unsigned)(size_t)(&Vs[0][0]);
#endif

  float mrow[8], lrow[8];
  v8f O[4] = {};
  #pragma unroll
  for (int r = 0; r < 8; ++r) { mrow[r] = -1e30f; lrow[r] = 0.0f; }

  for (int j0 = 0; j0 < i0 + 16; j0 += 32) {
    // ---- V tile -> LDS ----------------------------------------------------
#if USE_TDM
    {
      // D# group0: count=1, lds_addr, 57-bit global addr, type=2 ("image")
      const unsigned long long ga =
          (unsigned long long)(size_t)(Vg + bhoff + (size_t)j0 * 64);
      u32x4 g0;
      g0[0] = 1u;
      g0[1] = lds_vs;
      g0[2] = (unsigned)(ga & 0xFFFFFFFFu);
      g0[3] = (unsigned)((ga >> 32) & 0x1FFFFFFu) | (2u << 30);
      // D# group1: data_size=8B, pad every 32 dwords by 4 dwords (row 128B->144B)
      // tensor_dim0=16 (8B units), tensor_dim1=4096, tile=16x32, stride0=16.
      i32x8 g1;
      g1[0] = (3 << 16) | (1 << 20) | (4 << 22) | (3 << 25);
      g1[1] = (16 << 16);            // tensor_dim0[15:0]
      g1[2] = (int)(4096u << 16);    // tensor_dim1[15:0]
      g1[3] = (16 << 16);            // tile_dim0
      g1[4] = 32;                    // tile_dim1
      g1[5] = 16;                    // tensor_dim0_stride[31:0]
      g1[6] = 0;
      g1[7] = 0;
      i32x4 z4 = {};
#if defined(__clang_major__) && (__clang_major__ >= 23)
      i32x8 z8 = {};
      __builtin_amdgcn_tensor_load_to_lds(g0, g1, z4, z4, z8, 0);
#else
      __builtin_amdgcn_tensor_load_to_lds(g0, g1, z4, z4, 0);
#endif
    }
#else
    {
      const int4* vsrc = (const int4*)(Vg + bhoff + (size_t)(j0 + lane) * 64);
      int4* vdst = (int4*)(&Vs[lane][0]);
      #pragma unroll
      for (int i = 0; i < 8; ++i) vdst[i] = vsrc[i];
    }
#endif
    // ---- Scores for two 16-key subtiles (overlaps with TDM) ---------------
    v8f s[2];
    #pragma unroll
    for (int sub = 0; sub < 2; ++sub) {
      const _Float16* krow =
          Kg + bhoff + (size_t)(j0 + sub * 16 + (lane & 15)) * 64 + (lane & 16);
      v16h kb0, kb1;
      #pragma unroll
      for (int e = 0; e < 16; ++e) { kb0[e] = krow[e]; kb1[e] = krow[32 + e]; }
      v8f acc = {};
      acc = WMMA_F16(qa0, kb0, acc);
      acc = WMMA_F16(qa1, kb1, acc);
      s[sub] = acc;
    }
    // Causal mask
    const int c0 = j0 + (lane & 15), c1 = c0 + 16;
    #pragma unroll
    for (int r = 0; r < 8; ++r) {
      const int row = i0 + r + ((lane & 16) >> 1);
      if (c0 > row) s[0][r] = -1e30f;
      if (c1 > row) s[1][r] = -1e30f;
    }
    __syncthreads();
    // Online softmax (rows live across 16 lanes of each half-wave)
    #pragma unroll
    for (int r = 0; r < 8; ++r) {
      float t = fmaxf(s[0][r], s[1][r]);
      #pragma unroll
      for (int m = 1; m < 16; m <<= 1) t = fmaxf(t, __shfl_xor(t, m, 16));
      const float mnew = fmaxf(mrow[r], t);
      const float corr = __expf(mrow[r] - mnew);
      mrow[r] = mnew;
      const float p0 = __expf(s[0][r] - mnew);
      const float p1 = __expf(s[1][r] - mnew);
      float psum = p0 + p1;
      #pragma unroll
      for (int m = 1; m < 16; m <<= 1) psum += __shfl_xor(psum, m, 16);
      lrow[r] = lrow[r] * corr + psum;
      #pragma unroll
      for (int d = 0; d < 4; ++d) O[d][r] *= corr;
      const int rm = r + ((lane & 16) >> 1);
      P[rm][lane & 15]        = (_Float16)p0;
      P[rm][16 + (lane & 15)] = (_Float16)p1;
    }
    __syncthreads();
#if USE_TDM
    __builtin_amdgcn_s_wait_tensorcnt(0);   // V tile resident before reads
#endif
    // ---- O += P @ V -------------------------------------------------------
    v16h pa;
    #pragma unroll
    for (int e = 0; e < 16; ++e) pa[e] = P[lane & 15][a_k(e, lane)];
    #pragma unroll
    for (int d = 0; d < 4; ++d) {
      v16h vb;
      #pragma unroll
      for (int e = 0; e < 16; ++e) vb[e] = Vs[(lane & 16) + e][d * 16 + (lane & 15)];
      O[d] = WMMA_F16(pa, vb, O[d]);
    }
    __syncthreads();   // DS fence: fragment reads done before next V DMA / P write
  }

  // Normalize and store to attn buffer [B*N, 1024] f16
  #pragma unroll
  for (int r = 0; r < 8; ++r) {
    const float inv = 1.0f / lrow[r];
    const int m = r + ((lane & 16) >> 1);
    const size_t base = ((size_t)(b * 2048 + i0 + m)) * 1024 + h * 64;
    #pragma unroll
    for (int d = 0; d < 4; ++d)
      attnb[base + d * 16 + (lane & 15)] = (_Float16)(O[d][r] * inv);
  }
}

// ---------------------------------------------------------------------------
// Output projection: [4096,1024] x [1024,1024]^T + bias -> f32 out
// ---------------------------------------------------------------------------
__global__ __launch_bounds__(256) void proj_gemm(
    const _Float16* __restrict__ A,   // attn [4096,1024]
    const _Float16* __restrict__ W,   // proj_w [1024,1024]
    const float*    __restrict__ bias,
    float* __restrict__ out) {
  constexpr int BM = 128, BN = 64, BK = 32, LP = 40;
  __shared__ __align__(16) _Float16 As[BM][LP];
  __shared__ __align__(16) _Float16 Bs[BN][LP];
  const int tid = threadIdx.x;
  const int lane = tid & 31, wave = tid >> 5;
  const int tm = blockIdx.x, tn = blockIdx.y;
  const int K = 1024;

  v8f acc[4] = {};

  for (int k0 = 0; k0 < K; k0 += BK) {
    __syncthreads();
    {
      const int row = tid >> 1, ks = (tid & 1) * 16;
      const _Float16* src = A + (size_t)(tm * BM + row) * K + k0 + ks;
      #pragma unroll
      for (int i = 0; i < 16; ++i) As[row][ks + i] = src[i];
      if (k0 + BK < K) __builtin_prefetch(src + BK, 0, 1);
    }
    {
      const int row = tid >> 2, ks = (tid & 3) * 8;
      const _Float16* src = W + (size_t)(tn * BN + row) * K + k0 + ks;
      #pragma unroll
      for (int i = 0; i < 8; ++i) Bs[row][ks + i] = src[i];
    }
    __syncthreads();

    v16h a;
    #pragma unroll
    for (int e = 0; e < 16; ++e) a[e] = As[wave * 16 + (lane & 15)][a_k(e, lane)];
    #pragma unroll
    for (int nt = 0; nt < 4; ++nt) {
      v16h b;
      #pragma unroll
      for (int e = 0; e < 16; ++e) b[e] = Bs[nt * 16 + (lane & 15)][b_k(e, lane)];
      acc[nt] = WMMA_F16(a, b, acc[nt]);
    }
  }

  #pragma unroll
  for (int nt = 0; nt < 4; ++nt) {
    const int j = tn * 64 + nt * 16 + (lane & 15);
    const float bj = bias[j];
    #pragma unroll
    for (int r = 0; r < 8; ++r) {
      const int mg = tm * 128 + wave * 16 + ((lane & 16) >> 1) + r;
      out[(size_t)mg * 1024 + j] = acc[nt][r] + bj;
    }
  }
}

// ---------------------------------------------------------------------------
// Launch
// ---------------------------------------------------------------------------
extern "C" void kernel_launch(void* const* d_in, const int* in_sizes, int n_in,
                              void* d_out, int out_size, void* d_ws, size_t ws_size,
                              hipStream_t stream) {
  const float* x      = (const float*)d_in[0];
  const float* qkv_w  = (const float*)d_in[1];
  const float* qkv_b  = (const float*)d_in[2];
  const float* proj_w = (const float*)d_in[3];
  const float* proj_b = (const float*)d_in[4];
  float* out = (float*)d_out;

  // Workspace layout (~49 MB)
  _Float16* xh     = (_Float16*)d_ws;
  _Float16* wqkvh  = xh + (size_t)4096 * 1024;
  _Float16* wprojh = wqkvh + (size_t)3072 * 1024;
  _Float16* Qg     = wprojh + (size_t)1024 * 1024;
  _Float16* Kg     = Qg + (size_t)32 * 2048 * 64;
  _Float16* Vg     = Kg + (size_t)32 * 2048 * 64;
  _Float16* attnb  = Vg + (size_t)32 * 2048 * 64;
  float*    ropeCS = (float*)(attnb + (size_t)4096 * 1024);  // 2*131072 floats

  const int nx = 4096 * 1024, nw = 3072 * 1024, np = 1024 * 1024, nt = 2048 * 64;
  cvt_f16<<<(nx + 255) / 256, 256, 0, stream>>>(x, xh, nx);
  cvt_f16<<<(nw + 255) / 256, 256, 0, stream>>>(qkv_w, wqkvh, nw);
  cvt_f16<<<(np + 255) / 256, 256, 0, stream>>>(proj_w, wprojh, np);
  rope_tables<<<(nt + 255) / 256, 256, 0, stream>>>(ropeCS);

  qkv_gemm_rope<<<dim3(32, 48), 256, 0, stream>>>(xh, wqkvh, qkv_b, ropeCS, Qg, Kg, Vg);
  attn_fwd<<<dim3(128, 32), 32, 0, stream>>>(Qg, Kg, Vg, attnb);
  proj_gemm<<<dim3(32, 16), 256, 0, stream>>>(attnb, wprojh, proj_b, out);
}
